// SetAbstraction_57930518888877
// MI455X (gfx1250) — compile-verified, matrix operand order checked by reading, not run on hardware
//
#include <hip/hip_runtime.h>
#include <stdint.h>

typedef _Float16 f16;
typedef __attribute__((ext_vector_type(16))) _Float16 v16h;
typedef __attribute__((ext_vector_type(8)))  _Float16 v8h;
typedef __attribute__((ext_vector_type(8)))  float    v8f;

#define BATCH 16
#define NPTS  4096
#define MCENT 1024
#define KNbr  32
#define COLS  32768          // MCENT * KNbr per batch
#define R2    0.04f          // RADIUS^2

// ---------------------------------------------------------------- FPS
// One block per batch, 512 threads (16 waves). Distances + coords in registers
// (8 pts/thread); coords mirrored in LDS so the reducing wave can fetch the
// winner's coordinates directly -> only 2 barriers per iteration.
__global__ __launch_bounds__(512) void fps_kernel(const float* __restrict__ xyz,
                                                  int* __restrict__ fps_idx)
{
    __shared__ float px[NPTS], py[NPTS], pz[NPTS];
    __shared__ float wval[16];
    __shared__ int   widx[16];
    __shared__ float bc[3];

    const int b = blockIdx.x;
    const int t = threadIdx.x;
    const int lane = t & 31, wid = t >> 5;
    const float* xb = xyz + (size_t)b * 3 * NPTS;

    float lx[8], ly[8], lz[8], ld[8];
    const float x0 = xb[0], y0 = xb[NPTS], z0 = xb[2 * NPTS];
#pragma unroll
    for (int j = 0; j < 8; ++j) {
        int n = t + j * 512;
        float x = xb[n], y = xb[NPTS + n], z = xb[2 * NPTS + n];
        lx[j] = x; ly[j] = y; lz[j] = z;
        px[n] = x; py[n] = y; pz[n] = z;
        float dx = x - x0, dy = y - y0, dz = z - z0;
        ld[j] = dx * dx + dy * dy + dz * dz;
    }
    if (t == 0) fps_idx[b * MCENT] = 0;
    __syncthreads();

    for (int it = 1; it < MCENT; ++it) {
        // local argmax (ascending n -> strict > keeps lowest index on ties)
        float bv = -1.f; int bi = 0;
#pragma unroll
        for (int j = 0; j < 8; ++j)
            if (ld[j] > bv) { bv = ld[j]; bi = t + j * 512; }
#pragma unroll
        for (int off = 16; off; off >>= 1) {
            float ov = __shfl_down(bv, off, 32);
            int   oi = __shfl_down(bi, off, 32);
            if (ov > bv || (ov == bv && oi < bi)) { bv = ov; bi = oi; }
        }
        if (lane == 0) { wval[wid] = bv; widx[wid] = bi; }
        __syncthreads();
        if (t < 32) {
            bv = (lane < 16) ? wval[lane] : -1.f;
            bi = (lane < 16) ? widx[lane] : 0;
#pragma unroll
            for (int off = 8; off; off >>= 1) {
                float ov = __shfl_down(bv, off, 32);
                int   oi = __shfl_down(bi, off, 32);
                if (ov > bv || (ov == bv && oi < bi)) { bv = ov; bi = oi; }
            }
            if (lane == 0) {
                fps_idx[b * MCENT + it] = bi;
                bc[0] = px[bi]; bc[1] = py[bi]; bc[2] = pz[bi];
            }
        }
        __syncthreads();
        float cx = bc[0], cy = bc[1], cz = bc[2];
#pragma unroll
        for (int j = 0; j < 8; ++j) {
            float dx = lx[j] - cx, dy = ly[j] - cy, dz = lz[j] - cz;
            float d2 = dx * dx + dy * dy + dz * dz;
            if (d2 < ld[j]) ld[j] = d2;
        }
    }
}

// ---------------------------------------------------------------- gather new_xyz
__global__ void gather_newxyz_kernel(const float* __restrict__ xyz,
                                     const int* __restrict__ fps_idx,
                                     float* __restrict__ out_xyz)
{
    int tid = blockIdx.x * blockDim.x + threadIdx.x;
    if (tid >= BATCH * 3 * MCENT) return;
    int m = tid % MCENT;
    int c = (tid / MCENT) % 3;
    int b = tid / (3 * MCENT);
    int n = fps_idx[b * MCENT + m];
    out_xyz[tid] = xyz[((size_t)b * 3 + c) * NPTS + n];
}

// ---------------------------------------------------------------- ball query
// One wave per centroid; lane l tests point n0 + l, hits compacted into slots by ballot prefix count (K == wave32 width).
__global__ __launch_bounds__(256) void ball_query_kernel(const float* __restrict__ xyz,
                                                         const float* __restrict__ new_xyz,
                                                         int* __restrict__ ball_idx)
{
    int gwave = (blockIdx.x * blockDim.x + threadIdx.x) >> 5;
    int lane  = threadIdx.x & 31;
    if (gwave >= BATCH * MCENT) return;
    int b = gwave / MCENT, m = gwave % MCENT;
    const float* xb = xyz + (size_t)b * 3 * NPTS;
    float cx = new_xyz[((size_t)b * 3 + 0) * MCENT + m];
    float cy = new_xyz[((size_t)b * 3 + 1) * MCENT + m];
    float cz = new_xyz[((size_t)b * 3 + 2) * MCENT + m];
    int* out = ball_idx + ((size_t)b * MCENT + m) * KNbr;

    int found = 0, first = 0;
    for (int n0 = 0; n0 < NPTS && found < KNbr; n0 += 32) {
        int n = n0 + lane;
        float dx = xb[n] - cx, dy = xb[NPTS + n] - cy, dz = xb[2 * NPTS + n] - cz;
        float d2 = dx * dx + dy * dy + dz * dz;
        unsigned mask = __builtin_amdgcn_ballot_w32(d2 < R2);
        if (mask && found == 0) first = n0 + __ffs(mask) - 1;
        if ((mask >> lane) & 1u) {
            int slot = found + __popc(mask & ((1u << lane) - 1u));
            if (slot < KNbr) out[slot] = n;
        }
        found += __popc(mask);
    }
    if (found < KNbr) {
        int f = found ? first : 0;           // no hit -> all zeros (matches ref)
        if (lane >= found) out[lane] = f;
    }
}

// ---------------------------------------------------------------- build grouped features
// One thread per (b,col): 96 channels written as 12 contiguous b128 stores.
__global__ __launch_bounds__(256) void build_x0_kernel(const float* __restrict__ feature,
                                                       const float* __restrict__ xyz,
                                                       const float* __restrict__ new_xyz,
                                                       const int* __restrict__ ball_idx,
                                                       f16* __restrict__ X)
{
    int tid = blockIdx.x * blockDim.x + threadIdx.x;
    if (tid >= BATCH * COLS) return;
    int b   = tid >> 15;          // / COLS
    int col = tid & (COLS - 1);
    int m = col >> 5, k = col & 31;
    int n = ball_idx[((size_t)b * MCENT + m) * KNbr + k];

    const float* fb = feature + (size_t)b * 64 * NPTS + n;
    f16 buf[96];
#pragma unroll
    for (int c = 0; c < 64; ++c) buf[c] = (f16)fb[(size_t)c * NPTS];
#pragma unroll
    for (int d = 0; d < 3; ++d)
        buf[64 + d] = (f16)(xyz[((size_t)b * 3 + d) * NPTS + n]
                            - new_xyz[((size_t)b * 3 + d) * MCENT + m]);
#pragma unroll
    for (int c = 67; c < 96; ++c) buf[c] = (f16)0.f;

    v8h* dst = (v8h*)(X + (size_t)tid * 96);
#pragma unroll
    for (int j = 0; j < 12; ++j) dst[j] = *(const v8h*)(buf + 8 * j);
}

// ---------------------------------------------------------------- weight pack f32 -> f16 (K padded)
__global__ void pack_w_kernel(const float* __restrict__ w, f16* __restrict__ wp,
                              int O, int Cin, int Cpad)
{
    int tid = blockIdx.x * blockDim.x + threadIdx.x;
    if (tid >= O * Cpad) return;
    int ck = tid % Cpad, o = tid / Cpad;
    wp[tid] = (f16)(ck < Cin ? w[o * Cin + ck] : 0.f);
}

__global__ void zero_stats_kernel(float* __restrict__ stats)
{
    stats[threadIdx.x] = 0.f;   // 256 threads: sum[128] + sumsq[128]
}

// ---------------------------------------------------------------- WMMA GEMM + fused BN statistics
// One wave per 16-column tile; ALL row tiles accumulated in registers so the
// activation matrix is read exactly once per layer. Stats: wave shfl-reduce ->
// LDS accumulate -> one set of global atomics per block.
template<int CIN, int ROWT>
__global__ __launch_bounds__(256) void gemm_kernel(const f16* __restrict__ X,
                                                   const f16* __restrict__ Wp,
                                                   f16* __restrict__ Y,
                                                   float* __restrict__ stats)
{
    constexpr int COUT = ROWT * 16;
    __shared__ float s_stat[256];
    const int t = threadIdx.x;
    s_stat[t] = 0.f;
    __syncthreads();

    const int lane  = t & 31;
    const int gwave = (blockIdx.x * 256 + t) >> 5;
    const int ct = gwave & 2047;                 // 2048 column tiles per batch
    const int b  = gwave >> 11;
    const int col0 = ct << 4;
    const int half = lane >> 4, l16 = lane & 15, kb = half * 8;

    const f16* xrow = X + ((size_t)b * COLS + col0 + l16) * CIN;   // B: col N = l16
    const f16* wr   = Wp + (size_t)l16 * CIN;                      // A: row M = l16

    v8f zero = {};
    v8f acc[ROWT];
#pragma unroll
    for (int rt = 0; rt < ROWT; ++rt) acc[rt] = zero;

#pragma unroll
    for (int kk = 0; kk < CIN; kk += 32) {
        v8h blo = *(const v8h*)(xrow + kk + kb);
        v8h bhi = *(const v8h*)(xrow + kk + 16 + kb);
        v16h bm;
#pragma unroll
        for (int e = 0; e < 8; ++e) { bm[e] = blo[e]; bm[e + 8] = bhi[e]; }
#pragma unroll
        for (int rt = 0; rt < ROWT; ++rt) {
            const f16* wrow = wr + (size_t)rt * 16 * CIN;
            v8h alo = *(const v8h*)(wrow + kk + kb);
            v8h ahi = *(const v8h*)(wrow + kk + 16 + kb);
            v16h a;
#pragma unroll
            for (int e = 0; e < 8; ++e) { a[e] = alo[e]; a[e + 8] = ahi[e]; }
            acc[rt] = __builtin_amdgcn_wmma_f32_16x16x32_f16(false, a, false, bm,
                                                             (short)0, acc[rt], false, false);
        }
    }

    // D layout: lane holds rows rt*16 + half*8 + r (r=0..7) of column col0+l16
    f16* yrow = Y + ((size_t)b * COLS + col0 + l16) * COUT + half * 8;
#pragma unroll
    for (int rt = 0; rt < ROWT; ++rt) {
        v8h dv; float s[8], q[8];
#pragma unroll
        for (int r = 0; r < 8; ++r) {
            float v = acc[rt][r];
            dv[r] = (f16)v; s[r] = v; q[r] = v * v;
        }
        *(v8h*)(yrow + rt * 16) = dv;
#pragma unroll
        for (int msk = 1; msk < 16; msk <<= 1)
#pragma unroll
            for (int r = 0; r < 8; ++r) {
                s[r] += __shfl_xor(s[r], msk, 32);
                q[r] += __shfl_xor(q[r], msk, 32);
            }
        if (l16 == 0) {
            int ch0 = rt * 16 + half * 8;
#pragma unroll
            for (int r = 0; r < 8; ++r) {
                atomicAdd(&s_stat[ch0 + r], s[r]);
                atomicAdd(&s_stat[128 + ch0 + r], q[r]);
            }
        }
    }
    __syncthreads();
    atomicAdd(&stats[t], s_stat[t]);
}

// ---------------------------------------------------------------- BN stats -> fused scale/bias
__global__ void bnparams_kernel(const float* __restrict__ stats,
                                const float* __restrict__ g,
                                const float* __restrict__ bb,
                                float* __restrict__ sb, int Cout)
{
    int o = blockIdx.x * blockDim.x + threadIdx.x;
    if (o >= Cout) return;
    const float cnt = (float)((long long)BATCH * COLS);
    float mean = stats[o] / cnt;
    float var  = stats[128 + o] / cnt - mean * mean;
    float sc   = g[o] * rsqrtf(var + 1e-5f);
    sb[o] = sc;
    sb[128 + o] = bb[o] - mean * sc;
}

// ---------------------------------------------------------------- BN+ReLU (Cout==64), v8h vectorized
__global__ __launch_bounds__(256) void apply_kernel(const f16* __restrict__ Y,
                                                    const float* __restrict__ sb,
                                                    f16* __restrict__ Xn)
{
    int tid = blockIdx.x * blockDim.x + threadIdx.x;      // B*COLS*8 octets
    if (tid >= BATCH * COLS * 8) return;
    int c0 = (tid & 7) << 3;
    v8h y = ((const v8h*)Y)[tid];
    v8h x;
#pragma unroll
    for (int j = 0; j < 8; ++j) {
        float v = (float)y[j] * sb[c0 + j] + sb[128 + c0 + j];
        x[j] = (f16)(v > 0.f ? v : 0.f);
    }
    ((v8h*)Xn)[tid] = x;
}

// ---------------------------------------------------------------- last layer: BN+ReLU+max over K neighbors
__global__ void final_max_kernel(const f16* __restrict__ Y2, const float* __restrict__ sb,
                                 float* __restrict__ out)
{
    int tid = blockIdx.x * blockDim.x + threadIdx.x;
    if (tid >= BATCH * 128 * MCENT) return;
    int o = tid & 127;                     // channel fastest -> lane-contiguous reads
    int m = (tid >> 7) & (MCENT - 1);
    int b = tid >> 17;
    const f16* y = Y2 + ((size_t)b * COLS + (size_t)m * KNbr) * 128 + o;
    float sc = sb[o], bi = sb[128 + o];
    float mx = 0.f;                        // relu(max) == max(0, max)
#pragma unroll 4
    for (int k = 0; k < KNbr; ++k) {
        float v = (float)y[(size_t)k * 128] * sc + bi;
        if (v > mx) mx = v;
    }
    out[((size_t)b * 128 + o) * MCENT + m] = mx;
}

// ================================================================ launch
extern "C" void kernel_launch(void* const* d_in, const int* in_sizes, int n_in,
                              void* d_out, int out_size, void* d_ws, size_t ws_size,
                              hipStream_t stream)
{
    const float* xyz     = (const float*)d_in[0];
    const float* feature = (const float*)d_in[1];
    const float* w0 = (const float*)d_in[2];
    const float* g0 = (const float*)d_in[3];
    const float* b0 = (const float*)d_in[4];
    const float* w1 = (const float*)d_in[5];
    const float* g1 = (const float*)d_in[6];
    const float* b1 = (const float*)d_in[7];
    const float* w2 = (const float*)d_in[8];
    const float* g2 = (const float*)d_in[9];
    const float* b2 = (const float*)d_in[10];

    char* ws = (char*)d_ws;
    int*   fps_idx  = (int*)ws;                                    //  64 KB
    int*   ball_idx = (int*)(ws + (64 << 10));                     //   2 MB
    f16*   wp0      = (f16*)(ws + 2162688);
    f16*   wp1      = (f16*)(ws + 2174976);
    f16*   wp2      = (f16*)(ws + 2183168);
    float* stats    = (float*)(ws + 2199552);                      // 256 f stats + 256 f scale/bias
    float* sb       = stats + 256;
    f16*   X        = (f16*)(ws + 2203648);                        //  96 MB (B*COLS*96 f16)
    f16*   Y        = (f16*)(ws + 2203648 + 100663296ll);          // 128 MB (B*COLS*128 f16)

    float* out_xyz  = (float*)d_out;                               // [B,3,M]
    float* out_feat = out_xyz + BATCH * 3 * MCENT;                 // [B,128,M]

    // 1. FPS + centroid gather
    fps_kernel<<<BATCH, 512, 0, stream>>>(xyz, fps_idx);
    gather_newxyz_kernel<<<(BATCH * 3 * MCENT + 255) / 256, 256, 0, stream>>>(xyz, fps_idx, out_xyz);

    // 2. Ball query (one wave32 per centroid)
    ball_query_kernel<<<BATCH * MCENT / 8, 256, 0, stream>>>(xyz, out_xyz, ball_idx);

    // 3. Grouped features -> f16, channel-contiguous, padded 67 -> 96
    build_x0_kernel<<<BATCH * COLS / 256, 256, 0, stream>>>(feature, xyz, out_xyz, ball_idx, X);

    // 4. Pack weights f32 -> f16 (pad K to multiple of 32)
    pack_w_kernel<<<(64 * 96 + 255) / 256, 256, 0, stream>>>(w0, wp0, 64, 67, 96);
    pack_w_kernel<<<(64 * 64 + 255) / 256, 256, 0, stream>>>(w1, wp1, 64, 64, 64);
    pack_w_kernel<<<(128 * 64 + 255) / 256, 256, 0, stream>>>(w2, wp2, 128, 64, 64);

    const int gemmBlocks  = BATCH * (COLS / 16) / 8;               // 4096 blocks (8 waves each)
    const int applyBlocks = BATCH * COLS * 8 / 256;                // v8h octets

    // layer 0: 96 -> 64
    zero_stats_kernel<<<1, 256, 0, stream>>>(stats);
    gemm_kernel<96, 4><<<gemmBlocks, 256, 0, stream>>>(X, wp0, Y, stats);
    bnparams_kernel<<<1, 64, 0, stream>>>(stats, g0, b0, sb, 64);
    apply_kernel<<<applyBlocks, 256, 0, stream>>>(Y, sb, X);

    // layer 1: 64 -> 64
    zero_stats_kernel<<<1, 256, 0, stream>>>(stats);
    gemm_kernel<64, 4><<<gemmBlocks, 256, 0, stream>>>(X, wp1, Y, stats);
    bnparams_kernel<<<1, 64, 0, stream>>>(stats, g1, b1, sb, 64);
    apply_kernel<<<applyBlocks, 256, 0, stream>>>(Y, sb, X);

    // layer 2: 64 -> 128, then BN+ReLU+max over neighbors
    zero_stats_kernel<<<1, 256, 0, stream>>>(stats);
    gemm_kernel<64, 8><<<gemmBlocks, 256, 0, stream>>>(X, wp2, Y, stats);
    bnparams_kernel<<<1, 128, 0, stream>>>(stats, g2, b2, sb, 128);
    final_max_kernel<<<(BATCH * 128 * MCENT + 255) / 256, 256, 0, stream>>>(Y, sb, out_feat);
}
